// LstmCell_28020366639121
// MI455X (gfx1250) — matrix operand order repaired and checked by reading, not compiled, not run on hardware
//
#include <hip/hip_runtime.h>
#include <hip/hip_bf16.h>
#include <stdint.h>

// ---------------------------------------------------------------------------
// Fused LSTM cell for MI455X (gfx1250, wave32, WMMA bf16).
// Pass 1: one-shot f32 -> bf16 conversion of x, h and the 6 weight matrices
//         into d_ws.
// Pass 2: fused 3-gate GEMM (128x64 block tile, 8 waves), bf16 WMMA with f32
//         accumulate. Tiles are staged into double-buffered LDS with
//         GLOBAL_LOAD_ASYNC_TO_LDS_B128 (zero staging VGPRs, ASYNCcnt-tracked
//         DMA overlapped with WMMAs), K-loop unrolled x2 so the LDS buffer
//         select is compile-time. Fused activation epilogue.
// ---------------------------------------------------------------------------

typedef __bf16 bf16x16 __attribute__((ext_vector_type(16)));
typedef float  f32x8   __attribute__((ext_vector_type(8)));
typedef unsigned short u16;

#define NB     8192   // batch rows
#define DK     2048   // D_IN == D_H
#define BM     128    // block tile rows
#define BN     64     // block tile cols (per gate)
#define BK     32     // K step == WMMA K
#define LDA    40     // padded LDS row stride (ushorts): 80B, 16B-aligned
#define KSTEPS 64     // K steps per phase (2048/32)
#define KBYTES (BK * 2)  // bytes of K advanced per step

union FragU {
    bf16x16 v;
    uint4   q[2];
};

__device__ __forceinline__ unsigned pack2_bf16(float a, float b) {
    unsigned ua = __float_as_uint(a);
    unsigned ub = __float_as_uint(b);
    ua += 0x7FFFu + ((ua >> 16) & 1u);   // round-to-nearest-even
    ub += 0x7FFFu + ((ub >> 16) & 1u);
    return (ua >> 16) | (ub & 0xFFFF0000u);
}

__device__ __forceinline__ float fast_sigmoid(float x) {
    return 1.0f / (1.0f + __expf(-x));
}
__device__ __forceinline__ float fast_tanh(float x) {
    return __builtin_fmaf(2.0f, fast_sigmoid(2.0f * x), -1.0f);
}

// async DMA: global (SGPR base + 32-bit lane byte-offset) -> LDS (lane addr)
#define ASYNC_B128(ldsa, voff, base)                                        \
    asm volatile("global_load_async_to_lds_b128 %0, %1, %2"                 \
                 :: "v"(ldsa), "v"(voff), "s"(base) : "memory")
#define WAIT_ASYNC0() asm volatile("s_wait_asynccnt 0x0" ::: "memory")

// ---------------- Pass 1: f32 -> bf16 conversion ---------------------------
__global__ __launch_bounds__(256) void cvt_bf16(const float4* __restrict__ src,
                                                uint2* __restrict__ dst, int n4) {
    int i = blockIdx.x * 256 + threadIdx.x;
    if (i < n4) {
        const float4 v = src[i];
        dst[i] = make_uint2(pack2_bf16(v.x, v.y), pack2_bf16(v.z, v.w));
    }
}

// ---------------- Pass 2: fused 3-gate GEMM + LSTM epilogue ----------------
__global__ __launch_bounds__(256) void lstm_fused_wmma(
    const u16* __restrict__ xb,  const u16* __restrict__ hb,
    const u16* __restrict__ w0,  const u16* __restrict__ w1,  // Wix, Wmx
    const u16* __restrict__ w2,  const u16* __restrict__ w3,  // Wox, Wih
    const u16* __restrict__ w4,  const u16* __restrict__ w5,  // Wmh, Woh
    const float* __restrict__ c,
    const float* __restrict__ bix, const float* __restrict__ bmx,
    const float* __restrict__ box_,
    const float* __restrict__ bih, const float* __restrict__ bmh,
    const float* __restrict__ boh,
    float* __restrict__ out_h,     float* __restrict__ out_c)
{
    __shared__ u16 As[2][BM][LDA];        // activations (x then h), bf16
    __shared__ u16 Bs[2][3][BN][LDA];     // 3 gate weight tiles, bf16

    const int t    = threadIdx.x;
    const int lane = t & 31;
    const int wave = t >> 5;
    const int wm   = wave & 3;             // wave's 32-row slot (0..3)
    const int wn   = wave >> 2;            // wave's 32-col slot (0..1)
    const int bm_base = blockIdx.x * BM;   // batch-row base
    const int j_base  = blockIdx.y * BN;   // gate-column base

    // DMA mapping: 256 threads, each issues 5 async b128 copies per K-step
    const int q  = t & 3;                  // 16B chunk within 32-col row
    const int r  = t >> 2;                 // row 0..63 (A also uses r+64)

    // per-lane fragment addressing (ISA 7.12.2)
    const int mrow  = lane & 15;           // M (A) / N (B) index within 16
    const int khalf = lane >> 4;           // K-half selector

    // global byte offsets (relative to each tensor base) for this lane's rows
    const unsigned goffA0 = ((unsigned)(bm_base + r)      * DK + q * 8) * 2u;
    const unsigned goffA1 = ((unsigned)(bm_base + r + 64) * DK + q * 8) * 2u;
    const unsigned goffW  = ((unsigned)(j_base + r)       * DK + q * 8) * 2u;

    // wave-relative LDS byte addresses (buffer 0); buffer 1 adds the stride
    const unsigned ldsA0 = (unsigned)(uintptr_t)&As[0][r][q * 8];
    const unsigned ldsA1 = (unsigned)(uintptr_t)&As[0][r + 64][q * 8];
    const unsigned ldsB0 = (unsigned)(uintptr_t)&Bs[0][0][r][q * 8];
    const unsigned ldsB1 = (unsigned)(uintptr_t)&Bs[0][1][r][q * 8];
    const unsigned ldsB2 = (unsigned)(uintptr_t)&Bs[0][2][r][q * 8];

    f32x8 acc[3][2][2];
    const f32x8 zero = {0.f,0.f,0.f,0.f,0.f,0.f,0.f,0.f};
    #pragma unroll
    for (int g = 0; g < 3; ++g)
        #pragma unroll
        for (int i = 0; i < 2; ++i)
            #pragma unroll
            for (int j = 0; j < 2; ++j)
                acc[g][i][j] = zero;

    // issue the 5 async copies for one 128x32 A tile + 3x 64x32 weight tiles
    auto issue_tile = [&](const u16* Ap, const u16* W0, const u16* W1,
                          const u16* W2, int buf /*literal*/, int kbyte) {
        const unsigned ao = (unsigned)(buf * (int)sizeof(As[0]));
        const unsigned bo = (unsigned)(buf * (int)sizeof(Bs[0]));
        ASYNC_B128(ldsA0 + ao, goffA0 + kbyte, Ap);
        ASYNC_B128(ldsA1 + ao, goffA1 + kbyte, Ap);
        ASYNC_B128(ldsB0 + bo, goffW + kbyte, W0);
        ASYNC_B128(ldsB1 + bo, goffW + kbyte, W1);
        ASYNC_B128(ldsB2 + bo, goffW + kbyte, W2);
    };

    // consume LDS buffer `cur` (literal): frag loads + 12 WMMAs
    auto compute = [&](int cur) {
        FragU afr[2];
        #pragma unroll
        for (int tm = 0; tm < 2; ++tm) {
            const int m = wm * 32 + tm * 16 + mrow;
            afr[tm].q[0] = *(const uint4*)&As[cur][m][khalf * 8];
            afr[tm].q[1] = *(const uint4*)&As[cur][m][16 + khalf * 8];
        }
        #pragma unroll
        for (int g = 0; g < 3; ++g) {
            FragU bfr[2];
            #pragma unroll
            for (int tn = 0; tn < 2; ++tn) {
                const int n = wn * 32 + tn * 16 + mrow;
                bfr[tn].q[0] = *(const uint4*)&Bs[cur][g][n][khalf * 8];
                bfr[tn].q[1] = *(const uint4*)&Bs[cur][g][n][16 + khalf * 8];
            }
            #pragma unroll
            for (int tm = 0; tm < 2; ++tm)
                #pragma unroll
                for (int tn = 0; tn < 2; ++tn)
                    acc[g][tm][tn] = __builtin_amdgcn_wmma_f32_16x16x32_bf16(
                        false, afr[tm].v, false, bfr[tn].v,
                        (short)0, acc[g][tm][tn], false, false);
        }
    };

    // ---- software-pipelined main loop, unrolled x2 (compile-time buffer) ---
    issue_tile(xb, w0, w1, w2, 0, 0);
    for (int kt = 0; kt < KSTEPS; kt += 2) {           // phase 0: x @ Wx^T
        WAIT_ASYNC0();                                 // tile kt landed (mine)
        __syncthreads();                               // everyone's landed
        issue_tile(xb, w0, w1, w2, 1, (kt + 1) * KBYTES);
        compute(0);
        WAIT_ASYNC0();
        __syncthreads();
        if (kt + 2 < KSTEPS)
            issue_tile(xb, w0, w1, w2, 0, (kt + 2) * KBYTES);
        else
            issue_tile(hb, w3, w4, w5, 0, 0);          // first tile of phase 1
        compute(1);
    }
    for (int kt = 0; kt < KSTEPS; kt += 2) {           // phase 1: h @ Wh^T
        WAIT_ASYNC0();
        __syncthreads();
        issue_tile(hb, w3, w4, w5, 1, (kt + 1) * KBYTES);
        compute(0);
        WAIT_ASYNC0();
        __syncthreads();
        if (kt + 2 < KSTEPS)
            issue_tile(hb, w3, w4, w5, 0, (kt + 2) * KBYTES);
        compute(1);
    }

    // ---- epilogue: bias + gates + state update ------------------------------
    // C layout: VGPR e -> M = e + 8*(lane>>4); N = lane&15
    const int nlo  = lane & 15;
    const int half = lane >> 4;
    #pragma unroll
    for (int tn = 0; tn < 2; ++tn) {
        const int col = j_base + wn * 32 + tn * 16 + nlo;
        const float bi = bix[col] + bih[col];
        const float bm = bmx[col] + bmh[col];
        const float bo = box_[col] + boh[col];
        #pragma unroll
        for (int tm = 0; tm < 2; ++tm) {
            #pragma unroll
            for (int e = 0; e < 8; ++e) {
                const int row = bm_base + wm * 32 + tm * 16 + e + 8 * half;
                const size_t idx = (size_t)row * DK + col;
                const float gi = acc[0][tm][tn][e] + bi;
                const float gm = acc[1][tm][tn][e] + bm;
                const float go = acc[2][tm][tn][e] + bo;
                const float is_ = fast_sigmoid(gi);
                const float fs  = fast_sigmoid(gm);
                const float ms  = fast_tanh(gm);
                const float os  = fast_sigmoid(go);
                const float cn  = __builtin_fmaf(fs, c[idx], is_ * ms);
                out_c[idx] = cn;
                out_h[idx] = os * fast_tanh(cn);
            }
        }
    }
}

extern "C" void kernel_launch(void* const* d_in, const int* in_sizes, int n_in,
                              void* d_out, int out_size, void* d_ws, size_t ws_size,
                              hipStream_t stream) {
    (void)in_sizes; (void)n_in; (void)out_size; (void)ws_size;
    const float* x    = (const float*)d_in[0];
    const float* h    = (const float*)d_in[1];
    const float* c    = (const float*)d_in[2];
    const float* Wix  = (const float*)d_in[3];
    const float* bix  = (const float*)d_in[4];
    const float* Wmx  = (const float*)d_in[5];
    const float* bmx  = (const float*)d_in[6];
    const float* Wox  = (const float*)d_in[7];
    const float* box_ = (const float*)d_in[8];
    const float* Wih  = (const float*)d_in[9];
    const float* bih  = (const float*)d_in[10];
    const float* Wmh  = (const float*)d_in[11];
    const float* bmh  = (const float*)d_in[12];
    const float* Woh  = (const float*)d_in[13];
    const float* boh  = (const float*)d_in[14];

    // workspace layout (bf16): xb, hb, Wix, Wmx, Wox, Wih, Wmh, Woh
    u16* xb = (u16*)d_ws;
    u16* hbuf = xb + (size_t)NB * DK;
    u16* wb = hbuf + (size_t)NB * DK;
    u16* w[6];
    for (int i = 0; i < 6; ++i) w[i] = wb + (size_t)i * DK * DK;

    // Pass 1: convert everything the GEMM consumes to bf16
    const int nAct4 = (NB * DK) / 4;     // x, h
    const int nW4   = (DK * DK) / 4;     // each weight matrix
    const float* wsrc[6] = { Wix, Wmx, Wox, Wih, Wmh, Woh };
    cvt_bf16<<<(nAct4 + 255) / 256, 256, 0, stream>>>((const float4*)x, (uint2*)xb, nAct4);
    cvt_bf16<<<(nAct4 + 255) / 256, 256, 0, stream>>>((const float4*)h, (uint2*)hbuf, nAct4);
    for (int i = 0; i < 6; ++i)
        cvt_bf16<<<(nW4 + 255) / 256, 256, 0, stream>>>((const float4*)wsrc[i], (uint2*)w[i], nW4);

    // Pass 2: fused GEMM + LSTM cell
    float* out_h = (float*)d_out;
    float* out_c = out_h + (size_t)NB * DK;
    dim3 grid(NB / BM, DK / BN);
    dim3 block(256);
    lstm_fused_wmma<<<grid, block, 0, stream>>>(
        xb, hbuf, w[0], w[1], w[2], w[3], w[4], w[5],
        c, bix, bmx, box_, bih, bmh, boh, out_h, out_c);
}
